// MNIST_TTM_27281632264656
// MI455X (gfx1250) — matrix operand order repaired and checked by reading, not compiled, not run on hardware
//
#include <hip/hip_runtime.h>

// ---------------------------------------------------------------------------
// MNIST TT-MLP on gfx1250: pre-contract TT cores -> dense W1 (784x256, padded
// K->800) and W2 (256x10, padded N->16) in bf16, then fused
// relu(X@W1)@W2 with v_wmma_f32_16x16x32_bf16.
// Single-wave workgroups: the Tensor Data Mover double-buffers the B operand
// into LDS (TENSORcnt ordering, zero barriers), with LDS padding chosen so
// ds_load_b128 B-fragment reads are bank-conflict-free.
// ---------------------------------------------------------------------------

typedef __attribute__((ext_vector_type(16))) __bf16 v16bf;
typedef __attribute__((ext_vector_type(8)))  float  v8f;

#define K1PAD 800      // padded K for layer 1 (784 -> 25 chunks of 32)
#define NCOL  256      // layer-1 output columns
#define CSTRIDE 80     // LDS bytes per staged W1^T column (64B data + 16B pad)
#define BUFSZ  (NCOL * CSTRIDE)   // 20480 B per chunk buffer
#define HSTRIDE 528    // LDS bytes per H row (512B data + 16B pad)

union Frag { uint4 u[2]; v16bf v; };   // 32 bytes = 8 VGPRs = one WMMA A/B operand

// ---- workspace layout (bytes) ---------------------------------------------
#define WS_W1T   0                         // bf16 [256][800]  = 409600
#define WS_W2T   409600                    // bf16 [16][256]   = 8192
#define WS_A1    417792                    // f32  7168        = 28672
#define WS_A2    446464                    // f32  1280        = 5120

#if defined(__has_builtin)
#if __has_builtin(__builtin_amdgcn_tensor_load_to_lds) && \
    __has_builtin(__builtin_amdgcn_s_wait_tensorcnt)
#define HAVE_TDM 1
#endif
#endif
#ifndef HAVE_TDM
#define HAVE_TDM 0
#endif

// ===========================================================================
// Stage-1 contraction: A1[m0,n0,m1,n1,r2] and A2[m0,m1,n1,r2]
// ===========================================================================
__global__ __launch_bounds__(256) void build_stage1(
    const float* __restrict__ c10, const float* __restrict__ c11,
    const float* __restrict__ c20, const float* __restrict__ c21,
    float* __restrict__ A1, float* __restrict__ A2)
{
    int id = blockIdx.x * 256 + threadIdx.x;
    if (id < 7168) {                       // A1: dims (4,4,7,4,16)
        int r2 = id & 15; int t = id >> 4;
        int n1 = t & 3;   t >>= 2;
        int m1 = t % 7;   t /= 7;
        int n0 = t & 3;   int m0 = t >> 2;
        float s = 0.f;
        #pragma unroll
        for (int r1 = 0; r1 < 16; ++r1)
            s += c10[(m0*4 + n0)*16 + r1] * c11[((r1*7 + m1)*4 + n1)*16 + r2];
        A1[id] = s;
    } else if (id < 7168 + 1280) {         // A2: dims (4,4,5,16)
        int id2 = id - 7168;
        int r2 = id2 & 15; int t = id2 >> 4;
        int n1 = t % 5;    t /= 5;
        int m1 = t & 3;    int m0 = t >> 2;
        float s = 0.f;
        #pragma unroll
        for (int r1 = 0; r1 < 16; ++r1)
            s += c20[m0*16 + r1] * c21[((r1*4 + m1)*5 + n1)*16 + r2];
        A2[id2] = s;
    }
}

// ===========================================================================
// W1^T build: W1t[j][i] = sum_{r3} (sum_{r2} A1[m0,n0,m1,n1,r2]*G2[r2,m2,n2,r3])
//                                  * G3[r3,m3,n3]
// ===========================================================================
__global__ __launch_bounds__(256) void build_w1t(
    const float* __restrict__ A1, const float* __restrict__ c12,
    const float* __restrict__ c13, __bf16* __restrict__ W1t)
{
    const int i = blockIdx.x;              // 0..783 row of W1
    const int j = threadIdx.x;             // 0..255 col of W1
    const int m0 = i / 196, m1 = (i % 196) / 28, m2 = (i % 28) / 4, m3 = i & 3;
    const int n0 = j >> 6, n1 = (j >> 4) & 3, n2 = (j >> 2) & 3, n3 = j & 3;
    const float* a = A1 + (((m0*4 + n0)*7 + m1)*4 + n1)*16;   // 16 r2 values
    float acc = 0.f;
    #pragma unroll
    for (int r3 = 0; r3 < 16; ++r3) {
        float s = 0.f;
        #pragma unroll
        for (int r2 = 0; r2 < 16; ++r2)
            s += a[r2] * c12[((r2*7 + m2)*4 + n2)*16 + r3];
        acc += s * c13[(r3*4 + m3)*4 + n3];
    }
    W1t[j * K1PAD + i] = (__bf16)acc;      // k>=784 stays zero (memset)
}

// ===========================================================================
// W2^T build: W2t[j][i], j = n1*2+n2 in [0,10)
// ===========================================================================
__global__ __launch_bounds__(256) void build_w2t(
    const float* __restrict__ A2, const float* __restrict__ c22,
    const float* __restrict__ c23, __bf16* __restrict__ W2t)
{
    const int j = blockIdx.x;              // 0..9
    const int i = threadIdx.x;             // 0..255
    const int m0 = i >> 6, m1 = (i >> 4) & 3, m2 = (i >> 2) & 3, m3 = i & 3;
    const int n1 = j >> 1, n2 = j & 1;
    const float* a = A2 + ((m0*4 + m1)*5 + n1)*16;
    float acc = 0.f;
    #pragma unroll
    for (int r3 = 0; r3 < 16; ++r3) {
        float s = 0.f;
        #pragma unroll
        for (int r2 = 0; r2 < 16; ++r2)
            s += a[r2] * c22[((r2*4 + m2)*2 + n2)*16 + r3];
        acc += s * c23[r3*4 + m3];
    }
    W2t[j * NCOL + i] = (__bf16)acc;       // cols 10..15 stay zero (memset)
}

// ===========================================================================
// Fused GEMM: out = relu(X @ W1) @ W2
// ===========================================================================
__device__ inline unsigned pk2(float a, float b) {
    union { __bf16 h[2]; unsigned u; } p;
    p.h[0] = (__bf16)a; p.h[1] = (__bf16)b;
    return p.u;
}

// 8 consecutive f32 K-values -> 8 packed bf16 (16B); zeros past K=784
__device__ inline uint4 load8x(const float* __restrict__ row, int k) {
    if (k >= 784) return make_uint4(0u, 0u, 0u, 0u);
    const float4* p = (const float4*)(row + k);
    float4 a = p[0], b = p[1];
    uint4 r;
    r.x = pk2(a.x, a.y); r.y = pk2(a.z, a.w);
    r.z = pk2(b.x, b.y); r.w = pk2(b.z, b.w);
    return r;
}

#if HAVE_TDM
typedef unsigned int u32;
typedef __attribute__((ext_vector_type(4))) unsigned int v4u_;
typedef __attribute__((ext_vector_type(4))) int          v4i_;
typedef __attribute__((ext_vector_type(8))) int          v8i_;

// Issue one TDM 2D tile load: 32 (K, contiguous bf16) x 256 (columns, stride
// 800 elem) from W1^T into LDS at lds_addr, with 4 DWORDs of LDS padding per
// 16 DWORDs (-> 80B column stride, conflict-free ds_load_b128 afterwards).
__device__ inline void tdm_chunk(u32 lds_addr, unsigned long long ga) {
    // D# group 0: count=1 | lds_addr | global_addr | type=2
    v4u_ g0 = { 1u,
                lds_addr,
                (u32)ga,
                ((u32)(ga >> 32) & 0x01FFFFFFu) | 0x80000000u };
    // D# group 1: data_size=2B, pad_enable, pad_interval=16dw, pad_amount=4dw,
    // tensor_dim0=800, tensor_dim1=256, tile_dim0=32, tile_dim1=256,
    // tensor_dim0_stride=800
    v8i_ g1 = { (int)((1u << 16) | (1u << 20) | (3u << 22) | (3u << 25)),
                (int)(800u << 16),      // tensor_dim0[15:0] in [63:48]
                (int)(256u << 16),      // tensor_dim1[15:0] in [95:80]
                (int)(32u  << 16),      // tile_dim0 in [127:112]
                256,                    // tile_dim1 in [143:128]
                800,                    // tensor_dim0_stride[31:0]
                0, 0 };
    v4i_ z4 = { 0, 0, 0, 0 };
#if __clang_major__ >= 23
    v8i_ z8 = { 0, 0, 0, 0, 0, 0, 0, 0 };
    __builtin_amdgcn_tensor_load_to_lds(g0, g1, z4, z4, z8, 0);
#else
    __builtin_amdgcn_tensor_load_to_lds(g0, g1, z4, z4, 0);
#endif
}
#endif

__global__ __launch_bounds__(32) void ttm_fused(
    const float*  __restrict__ X,
    const __bf16* __restrict__ W1t,
    const __bf16* __restrict__ W2t,
    float*        __restrict__ out)
{
    // 2 x 20.5KB padded W1^T chunk buffers; later reused as H (16 x 528B)
    __shared__ __align__(16) unsigned char smem[2 * BUFSZ];

    const int lane = threadIdx.x & 31;
    const int hi16 = lane >> 4;            // 0: lanes 0-15, 1: lanes 16-31
    const int l16  = lane & 15;
    const int blockRow = blockIdx.x * 16;  // 512 blocks x 16 rows = 8192
    const float* xrow = X + (blockRow + l16) * 784;

    v8f zero8 = {};
    v8f acc[16];
    #pragma unroll
    for (int t = 0; t < 16; ++t) acc[t] = zero8;

#if HAVE_TDM
    const u32 lds0 = (u32)(uintptr_t)(void*)smem;
    const unsigned long long w1a = (unsigned long long)(uintptr_t)W1t;
    tdm_chunk(lds0, w1a);                  // chunk 0 -> buf 0 (per-wave DMA)
#else
    // cooperative fallback within the single wave: same padded layout
    auto stage = [&](int c, int buf) {
        const int kb = c * 32;
        unsigned char* b2 = smem + buf * BUFSZ;
        #pragma unroll
        for (int s = 0; s < 8; ++s) {
            const int col = lane + s * 32;
            const uint4* g = (const uint4*)(W1t + col * K1PAD + kb);
            uint4* l = (uint4*)(b2 + col * CSTRIDE);
            l[0] = g[0]; l[1] = g[1]; l[2] = g[2]; l[3] = g[3];
        }
    };
    stage(0, 0);
#endif

    // 16 WMMAs over one staged chunk, 2-slot pipelined B-fragment loads
    auto compute16 = [&](const unsigned char* base, const Frag& a) {
        Frag b[2];
        {
            const uint4* l = (const uint4*)(base + l16 * CSTRIDE + hi16 * 16);
            b[0].u[0] = l[0]; b[0].u[1] = l[2];
        }
        #pragma unroll
        for (int t = 0; t < 16; ++t) {
            if (t + 1 < 16) {
                const uint4* l = (const uint4*)
                    (base + ((t + 1) * 16 + l16) * CSTRIDE + hi16 * 16);
                b[(t + 1) & 1].u[0] = l[0];
                b[(t + 1) & 1].u[1] = l[2];
            }
            acc[t] = __builtin_amdgcn_wmma_f32_16x16x32_bf16(
                false, a.v, false, b[t & 1].v, (short)0, acc[t], false, false);
        }
    };

    // A fragment for chunk 0 (16x32 bf16: lanes<16 K 0-7/16-23)
    Frag a_cur, a_nxt;
    a_cur.u[0] = load8x(xrow, hi16 * 8);
    a_cur.u[1] = load8x(xrow, 16 + hi16 * 8);

    for (int c = 0; c < 24; ++c) {
        const int kbn = (c + 1) * 32;
        a_nxt.u[0] = load8x(xrow, kbn + hi16 * 8);        // prefetch next A
        a_nxt.u[1] = load8x(xrow, kbn + 16 + hi16 * 8);
#if HAVE_TDM
        tdm_chunk(lds0 + ((c + 1) & 1) * BUFSZ, w1a + (c + 1) * 64ull);
        // TDM completes in order: <=1 outstanding  =>  chunk c has landed
        __builtin_amdgcn_s_wait_tensorcnt(1);
#else
        stage(c + 1, (c + 1) & 1);
#endif
        compute16(smem + (c & 1) * BUFSZ, a_cur);
        a_cur = a_nxt;
    }
#if HAVE_TDM
    __builtin_amdgcn_s_wait_tensorcnt(0);
#endif
    compute16(smem + (24 & 1) * BUFSZ, a_cur);

    // ReLU, spill H to LDS as bf16 (16 rows x 256 cols, padded 528B stride);
    // single wave: LDS ops are in-order, no barrier needed
    unsigned char* h = smem;
    #pragma unroll
    for (int t = 0; t < 16; ++t) {
        const int col = t * 16 + l16;
        #pragma unroll
        for (int g = 0; g < 8; ++g) {
            const int r = hi16 * 8 + g;    // C/D layout: M = hi16*8+g
            float v = acc[t][g];
            v = v > 0.f ? v : 0.f;
            *((__bf16*)(h + r * HSTRIDE + col * 2)) = (__bf16)v;
        }
    }

    // layer 2: 16 rows x 16 cols (10 valid), K = 256 in 8 chunks, 2-slot pipe
    v8f acc2 = zero8;
    const unsigned char* hrow = h + l16 * HSTRIDE;
    Frag a2[2], b2[2];
    a2[0].u[0] = *(const uint4*)(hrow + (hi16 * 8) * 2);
    a2[0].u[1] = *(const uint4*)(hrow + (16 + hi16 * 8) * 2);
    {
        const uint4* pb = (const uint4*)(W2t + l16 * NCOL + hi16 * 8);
        b2[0].u[0] = pb[0]; b2[0].u[1] = pb[2];
    }
    #pragma unroll
    for (int kc = 0; kc < 8; ++kc) {
        if (kc + 1 < 8) {
            const int kb = (kc + 1) * 32;
            a2[(kc + 1) & 1].u[0] = *(const uint4*)(hrow + (kb + hi16 * 8) * 2);
            a2[(kc + 1) & 1].u[1] = *(const uint4*)(hrow + (kb + 16 + hi16 * 8) * 2);
            const uint4* pb = (const uint4*)(W2t + l16 * NCOL + kb + hi16 * 8);
            b2[(kc + 1) & 1].u[0] = pb[0];
            b2[(kc + 1) & 1].u[1] = pb[2];
        }
        acc2 = __builtin_amdgcn_wmma_f32_16x16x32_bf16(
            false, a2[kc & 1].v, false, b2[kc & 1].v, (short)0, acc2,
            false, false);
    }

    #pragma unroll
    for (int g = 0; g < 8; ++g) {
        const int r = blockRow + hi16 * 8 + g;
        if (l16 < 10) out[r * 10 + l16] = acc2[g];
    }
}

// ===========================================================================
extern "C" void kernel_launch(void* const* d_in, const int* in_sizes, int n_in,
                              void* d_out, int out_size, void* d_ws, size_t ws_size,
                              hipStream_t stream)
{
    const float* x   = (const float*)d_in[0];
    const float* c10 = (const float*)d_in[1];
    const float* c11 = (const float*)d_in[2];
    const float* c12 = (const float*)d_in[3];
    const float* c13 = (const float*)d_in[4];
    const float* c20 = (const float*)d_in[5];
    const float* c21 = (const float*)d_in[6];
    const float* c22 = (const float*)d_in[7];
    const float* c23 = (const float*)d_in[8];

    unsigned char* ws = (unsigned char*)d_ws;
    __bf16* W1t = (__bf16*)(ws + WS_W1T);
    __bf16* W2t = (__bf16*)(ws + WS_W2T);
    float*  A1  = (float*)(ws + WS_A1);
    float*  A2  = (float*)(ws + WS_A2);

    // zero padded weight buffers (capture-safe async op)
    hipMemsetAsync(ws, 0, WS_A1, stream);

    build_stage1<<<33, 256, 0, stream>>>(c10, c11, c20, c21, A1, A2);
    build_w1t<<<784, 256, 0, stream>>>(A1, c12, c13, W1t);
    build_w2t<<<10, 256, 0, stream>>>(A2, c22, c23, W2t);
    ttm_fused<<<512, 32, 0, stream>>>(x, W1t, W2t, (float*)d_out);
}
